// ECGformer_21629455302997
// MI455X (gfx1250) — matrix operand order repaired and checked by reading, not compiled, not run on hardware
//
#include <hip/hip_runtime.h>
#include <hip/hip_bf16.h>
#include <math.h>

typedef _Float16 f16;
typedef __attribute__((ext_vector_type(16))) _Float16 v16h;
typedef __attribute__((ext_vector_type(8)))  _Float16 v8h;
typedef __attribute__((ext_vector_type(8)))  float    v8f;
typedef __attribute__((ext_vector_type(4)))  int      v4i;

#define BB    8
#define NN    1024
#define CC    12
#define EE    256
#define HH    8
#define LL    4
#define XX    4
#define NCLS  5
#define TT    (NN + 1)        // 1025 tokens
#define MTOT  (BB * TT)       // 8200 rows
#define DHEAD 32
#define TPAD  1056            // T padded to multiple of 32
#define QT    ((TT + 15) / 16)  // 65 query/key tiles

union U16 { v16h v; v8h h2[2]; };

__device__ __forceinline__ float gelu_exact(float v) {
    return 0.5f * v * (1.0f + erff(v * 0.70710678118654752f));
}

__device__ __forceinline__ v8f wmma_f16(const U16& a, const U16& b, v8f c) {
    return __builtin_amdgcn_wmma_f32_16x16x32_f16(false, a.v, false, b.v,
                                                  (short)0, c, false, false);
}

// ---------------------------------------------------------------------------
// Async global -> LDS copy of 16B per lane (CDNA5 GLOBAL_LOAD_ASYNC_TO_LDS_B128)
// Builtin signature (from compiler diagnostic): (AS1 v4i*, AS3 v4i*, imm, imm)
// ---------------------------------------------------------------------------
__device__ __forceinline__ void async_cp16(const f16* gsrc, f16* ldst) {
#if defined(__has_builtin) && __has_builtin(__builtin_amdgcn_global_load_async_to_lds_b128)
    __builtin_amdgcn_global_load_async_to_lds_b128(
        (__attribute__((address_space(1))) v4i*)(void*)gsrc,
        (__attribute__((address_space(3))) v4i*)(void*)ldst, 0, 0);
#else
    *(v8h*)ldst = *(const v8h*)gsrc;
#endif
}

__device__ __forceinline__ void wait_async0() {
#if defined(__has_builtin) && __has_builtin(__builtin_amdgcn_s_wait_asynccnt)
    __builtin_amdgcn_s_wait_asynccnt(0);
#else
    asm volatile("s_wait_asynccnt 0" ::: "memory");
#endif
}

// ---------------------------------------------------------------------------
// Weight convert + transpose: Wt[n*K + k] = (f16) W[k*N + n]
// ---------------------------------------------------------------------------
__global__ void wt_k(const float* __restrict__ W, f16* __restrict__ Wt, int K, int N) {
    int idx = blockIdx.x * blockDim.x + threadIdx.x;
    if (idx >= K * N) return;
    int kk = idx / N, n = idx % N;
    Wt[(size_t)n * K + kk] = (f16)W[idx];
}

// ---------------------------------------------------------------------------
// Embedding: h[b,0,:] = cls ; h[b,1+n,:] = gelu(LN(x @ W_emb + b_emb))
// ---------------------------------------------------------------------------
__global__ void embed_k(const float* __restrict__ x, const float* __restrict__ W,
                        const float* __restrict__ be, const float* __restrict__ g,
                        const float* __restrict__ bb, const float* __restrict__ cls,
                        float* __restrict__ h) {
    __shared__ float xr[CC];
    __shared__ float red[EE];
    int b = blockIdx.x / TT;
    int t = blockIdx.x % TT;
    int e = threadIdx.x;
    size_t hidx = ((size_t)b * TT + t) * EE + e;
    if (t == 0) { h[hidx] = cls[e]; return; }
    int n = t - 1;
    if (e < CC) xr[e] = x[((size_t)b * NN + n) * CC + e];
    __syncthreads();
    float acc = be[e];
#pragma unroll
    for (int c = 0; c < CC; ++c) acc += xr[c] * W[c * EE + e];
    red[e] = acc; __syncthreads();
    for (int s = 128; s > 0; s >>= 1) { if (e < s) red[e] += red[e + s]; __syncthreads(); }
    float mean = red[0] / EE; __syncthreads();
    float d = acc - mean;
    red[e] = d * d; __syncthreads();
    for (int s = 128; s > 0; s >>= 1) { if (e < s) red[e] += red[e + s]; __syncthreads(); }
    float var = red[0] / EE;
    float o = d * rsqrtf(var + 1e-5f) * g[e] + bb[e];
    h[hidx] = gelu_exact(o);
}

// ---------------------------------------------------------------------------
// LayerNorm (optionally h += pos written back); outputs f16 y for WMMA GEMMs
// ---------------------------------------------------------------------------
__global__ void ln_k(float* __restrict__ h, const float* __restrict__ pos,
                     const float* __restrict__ g, const float* __restrict__ bta,
                     f16* __restrict__ y) {
    __shared__ float red[EE];
    int row = blockIdx.x;
    int e = threadIdx.x;
    int t = row % TT;
    size_t idx = (size_t)row * EE + e;
    float v = h[idx];
    if (pos) { v += pos[(size_t)t * EE + e]; h[idx] = v; }
    red[e] = v; __syncthreads();
    for (int s = 128; s > 0; s >>= 1) { if (e < s) red[e] += red[e + s]; __syncthreads(); }
    float mean = red[0] / EE; __syncthreads();
    float d = v - mean;
    red[e] = d * d; __syncthreads();
    for (int s = 128; s > 0; s >>= 1) { if (e < s) red[e] += red[e + s]; __syncthreads(); }
    float var = red[0] / EE;
    y[idx] = (f16)(d * rsqrtf(var + 1e-5f) * g[e] + bta[e]);
}

// ---------------------------------------------------------------------------
// WMMA GEMM: C[M,N] = A[M,K] (f16) @ Wt[N,K]^T (f16) + bias, epilogue modes:
//   0: f16 out           1: f32 out = resid + acc + bias (residual stream)
//   2: f16 out = gelu    3: f16 out + transposed copy into vT[B,H,32,TPAD]
// Block = 256 thr = 8 waves; tile 64(M) x 64(N); K stepped by 32.
// Double-buffered LDS staged by GLOBAL_LOAD_ASYNC_TO_LDS_B128; each wave
// computes a 16x32 strip: A fragment reused by two wmmas per K step.
// ---------------------------------------------------------------------------
__global__ void gemm_wmma(const f16* __restrict__ A, const f16* __restrict__ Wt,
                          const float* __restrict__ bias, void* __restrict__ outp,
                          float* __restrict__ resid, f16* __restrict__ vT,
                          int M, int N, int K, int mode) {
    __shared__ f16 As[2][64][40];  // +8 halves pad: 80B row stride, 16B aligned
    __shared__ f16 Bs[2][64][40];
    int tid  = threadIdx.x;
    int wave = tid >> 5, lane = tid & 31;
    int wm = wave >> 1, wn = wave & 1;       // 4(M) x 2(N) wave grid
    int blockM = blockIdx.x * 64;
    int blockN = blockIdx.y * 64;

    // staging coordinates: each thread moves 16B of A and 16B of B per K step
    int srow = tid >> 2;             // 0..63
    int scol = (tid & 3) * 8;        // 0,8,16,24
    int am = blockM + srow; if (am >= M) am = M - 1;
    const f16* aSrc = A  + (size_t)am * K + scol;
    const f16* bSrc = Wt + (size_t)(blockN + srow) * K + scol;

    // fragment coordinates
    int ar  = wm * 16 + (lane & 15);
    int klo = (lane < 16) ? 0 : 8;
    int br0 = wn * 32 + (lane & 15);
    int kb  = (lane >> 4) * 8;

    v8f acc0 = {}, acc1 = {};
    const int nk = K / 32;

    async_cp16(aSrc, &As[0][srow][scol]);
    async_cp16(bSrc, &Bs[0][srow][scol]);
    wait_async0();
    __syncthreads();

    int buf = 0;
    for (int s = 1; s <= nk; ++s) {
        if (s < nk) {   // prefetch next K step into the other buffer
            async_cp16(aSrc + s * 32, &As[buf ^ 1][srow][scol]);
            async_cp16(bSrc + s * 32, &Bs[buf ^ 1][srow][scol]);
        }
        U16 a, b0, b1;
        a.h2[0]  = *(const v8h*)&As[buf][ar][klo];
        a.h2[1]  = *(const v8h*)&As[buf][ar][16 + klo];
        b0.h2[0] = *(const v8h*)&Bs[buf][br0][kb];
        b0.h2[1] = *(const v8h*)&Bs[buf][br0][kb + 16];
        b1.h2[0] = *(const v8h*)&Bs[buf][br0 + 16][kb];
        b1.h2[1] = *(const v8h*)&Bs[buf][br0 + 16][kb + 16];
        acc0 = wmma_f16(a, b0, acc0);
        acc1 = wmma_f16(a, b1, acc1);
        if (s < nk) {
            wait_async0();
            __syncthreads();
            buf ^= 1;
        }
    }

    auto epi = [&](v8f acc, int ncol) {
        float bvv = bias ? bias[ncol] : 0.0f;
#pragma unroll
        for (int r = 0; r < 8; ++r) {
            int m = blockM + wm * 16 + ((lane < 16) ? r : (8 + r));
            if (m >= M) continue;
            float v = acc[r] + bvv;
            size_t oi = (size_t)m * N + ncol;
            if (mode == 1) {
                ((float*)outp)[oi] = resid[oi] + v;
            } else {
                if (mode == 2) v = gelu_exact(v);
                f16 hv = (f16)v;
                ((f16*)outp)[oi] = hv;
                if (mode == 3) {
                    int bidx = m / TT, t = m % TT;
                    vT[(((size_t)bidx * HH + (ncol >> 5)) * DHEAD + (ncol & 31)) * TPAD + t] = hv;
                }
            }
        }
    };
    epi(acc0, blockN + wn * 32 + (lane & 15));
    epi(acc1, blockN + wn * 32 + 16 + (lane & 15));
}

// ---------------------------------------------------------------------------
// Fused attention per (batch, head, 16-query tile). One wave per block.
// Pass 1: online rowmax/rowsum via QK^T wmma (head dim 32 == one K step).
// Pass 2: recompute scores, restage probs f16 in LDS, P@V via wmma with vT.
// Scale-after-softmax (/sqrt(E)=16) folded into output normalization.
// ---------------------------------------------------------------------------
__global__ void attn_wmma(const f16* __restrict__ q, const f16* __restrict__ k,
                          const f16* __restrict__ vT, f16* __restrict__ o) {
    __shared__ f16 pr[16][40];
    int lane = threadIdx.x;
    int qt = blockIdx.x % QT;
    int hh = (blockIdx.x / QT) % HH;
    int b  = blockIdx.x / (QT * HH);

    int mrow = lane & 15;
    int kcol = lane & 15;
    int klo  = (lane < 16) ? 0 : 8;
    int kb16 = (lane >> 4) * 8;

    int tq = qt * 16 + mrow; if (tq >= TT) tq = TT - 1;
    const f16* qrow = q + ((size_t)(b * TT + tq) * EE + hh * DHEAD);
    U16 af;
    af.h2[0] = *(const v8h*)(qrow + klo);
    af.h2[1] = *(const v8h*)(qrow + 16 + klo);

    float rmax[8], rsum[8];
#pragma unroll
    for (int r = 0; r < 8; ++r) { rmax[r] = -INFINITY; rsum[r] = 0.0f; }

    // Pass 1: online softmax statistics
    for (int kt = 0; kt < QT; ++kt) {
        int tk = kt * 16 + kcol;
        bool valid = tk < TT;
        int tkc = valid ? tk : TT - 1;
        const f16* krow = k + ((size_t)(b * TT + tkc) * EE + hh * DHEAD);
        U16 bf;
        bf.h2[0] = *(const v8h*)(krow + kb16);
        bf.h2[1] = *(const v8h*)(krow + kb16 + 16);
        v8f s = {};
        s = wmma_f16(af, bf, s);
#pragma unroll
        for (int r = 0; r < 8; ++r) {
            float sv = valid ? s[r] : -INFINITY;
            float mx = sv;  // half-wave (16-lane) reduction: rows live per half
            mx = fmaxf(mx, __shfl_xor(mx, 1));
            mx = fmaxf(mx, __shfl_xor(mx, 2));
            mx = fmaxf(mx, __shfl_xor(mx, 4));
            mx = fmaxf(mx, __shfl_xor(mx, 8));
            float nm = fmaxf(rmax[r], mx);
            float e = expf(sv - nm);
            e += __shfl_xor(e, 1); e += __shfl_xor(e, 2);
            e += __shfl_xor(e, 4); e += __shfl_xor(e, 8);
            rsum[r] = rsum[r] * expf(rmax[r] - nm) + e;
            rmax[r] = nm;
        }
    }

    // Pass 2: P @ V over 32-key chunks
    v8f oa = {}, ob = {};
    for (int kp = 0; kp < TPAD / 32; ++kp) {
#pragma unroll
        for (int half = 0; half < 2; ++half) {
            int kt = kp * 2 + half;
            int tk = kt * 16 + kcol;
            bool valid = tk < TT;
            int tkc = valid ? tk : TT - 1;
            const f16* krow = k + ((size_t)(b * TT + tkc) * EE + hh * DHEAD);
            U16 bf;
            bf.h2[0] = *(const v8h*)(krow + kb16);
            bf.h2[1] = *(const v8h*)(krow + kb16 + 16);
            v8f s = {};
            s = wmma_f16(af, bf, s);
#pragma unroll
            for (int r = 0; r < 8; ++r) {
                float sv = valid ? s[r] : -INFINITY;
                int m = (lane < 16) ? r : (8 + r);   // C-layout row of this lane
                pr[m][half * 16 + kcol] = (f16)expf(sv - rmax[r]);
            }
        }
        __syncthreads();
        U16 pf;  // restage probs as A-fragment (16 queries x 32 keys)
        pf.h2[0] = *(const v8h*)&pr[mrow][klo];
        pf.h2[1] = *(const v8h*)&pr[mrow][16 + klo];
        __syncthreads();
        int kbase = kp * 32 + kb16;
        const f16* v0 = vT + (((size_t)b * HH + hh) * DHEAD + kcol) * TPAD + kbase;
        const f16* v1 = vT + (((size_t)b * HH + hh) * DHEAD + 16 + kcol) * TPAD + kbase;
        U16 b0, b1;
        b0.h2[0] = *(const v8h*)v0; b0.h2[1] = *(const v8h*)(v0 + 16);
        b1.h2[0] = *(const v8h*)v1; b1.h2[1] = *(const v8h*)(v1 + 16);
        oa = wmma_f16(pf, b0, oa);
        ob = wmma_f16(pf, b1, ob);
    }

#pragma unroll
    for (int r = 0; r < 8; ++r) {
        int mr = (lane < 16) ? r : (8 + r);
        int t = qt * 16 + mr;
        if (t >= TT) continue;
        float inv = 1.0f / (rsum[r] * 16.0f);   // softmax norm * (1/sqrt(E))
        f16* orow = o + ((size_t)(b * TT + t) * EE + hh * DHEAD);
        orow[kcol]      = (f16)(oa[r] * inv);
        orow[16 + kcol] = (f16)(ob[r] * inv);
    }
}

// ---------------------------------------------------------------------------
// Classifier: mean-pool -> Linear -> LN -> Linear (tiny, scalar f32)
// ---------------------------------------------------------------------------
__global__ void cls_k(const float* __restrict__ h, const float* __restrict__ Wc1,
                      const float* __restrict__ bc1, const float* __restrict__ g,
                      const float* __restrict__ bb, const float* __restrict__ Wc2,
                      const float* __restrict__ bc2, float* __restrict__ out) {
    __shared__ float p[EE];
    __shared__ float red[EE];
    __shared__ float lno[EE];
    int b = blockIdx.x, e = threadIdx.x;
    float s = 0.0f;
    for (int t = 0; t < TT; ++t) s += h[((size_t)b * TT + t) * EE + e];
    p[e] = s / (float)TT;
    __syncthreads();
    float acc = bc1[e];
    for (int kx = 0; kx < EE; ++kx) acc += p[kx] * Wc1[kx * EE + e];
    red[e] = acc; __syncthreads();
    for (int ss = 128; ss > 0; ss >>= 1) { if (e < ss) red[e] += red[e + ss]; __syncthreads(); }
    float mean = red[0] / EE; __syncthreads();
    float d = acc - mean;
    red[e] = d * d; __syncthreads();
    for (int ss = 128; ss > 0; ss >>= 1) { if (e < ss) red[e] += red[e + ss]; __syncthreads(); }
    float var = red[0] / EE;
    lno[e] = d * rsqrtf(var + 1e-5f) * g[e] + bb[e];
    __syncthreads();
    if (e < NCLS) {
        float o = bc2[e];
        for (int kx = 0; kx < EE; ++kx) o += lno[kx] * Wc2[kx * NCLS + e];
        out[b * NCLS + e] = o;
    }
}

// ---------------------------------------------------------------------------
extern "C" void kernel_launch(void* const* d_in, const int* in_sizes, int n_in,
                              void* d_out, int out_size, void* d_ws, size_t ws_size,
                              hipStream_t stream) {
    (void)in_sizes; (void)n_in; (void)out_size; (void)ws_size;
    const float* x      = (const float*)d_in[0];
    const float* W_emb  = (const float*)d_in[1];
    const float* b_emb  = (const float*)d_in[2];
    const float* g_emb  = (const float*)d_in[3];
    const float* be_emb = (const float*)d_in[4];
    const float* clsT   = (const float*)d_in[5];
    const float* pos    = (const float*)d_in[6];
    const float* ln1_g  = (const float*)d_in[7];
    const float* ln1_b  = (const float*)d_in[8];
    const float* Wq     = (const float*)d_in[9];
    const float* bq     = (const float*)d_in[10];
    const float* Wk     = (const float*)d_in[11];
    const float* bk     = (const float*)d_in[12];
    const float* Wv     = (const float*)d_in[13];
    const float* bv_p   = (const float*)d_in[14];
    const float* Wo     = (const float*)d_in[15];
    const float* bo     = (const float*)d_in[16];
    const float* ln2_g  = (const float*)d_in[17];
    const float* ln2_b  = (const float*)d_in[18];
    const float* W1     = (const float*)d_in[19];
    const float* b1     = (const float*)d_in[20];
    const float* W2     = (const float*)d_in[21];
    const float* b2     = (const float*)d_in[22];
    const float* Wc1    = (const float*)d_in[23];
    const float* bc1    = (const float*)d_in[24];
    const float* lnc_g  = (const float*)d_in[25];
    const float* lnc_b  = (const float*)d_in[26];
    const float* Wc2    = (const float*)d_in[27];
    const float* bc2    = (const float*)d_in[28];

    char* ws = (char*)d_ws;
    size_t off = 0;
    auto alloc = [&](size_t bytes) -> void* {
        void* p = ws + off;
        off += (bytes + 255) & ~(size_t)255;
        return p;
    };
    float* h  = (float*)alloc((size_t)MTOT * EE * 4);
    f16* y    = (f16*)alloc((size_t)MTOT * EE * 2);
    f16* qb   = (f16*)alloc((size_t)MTOT * EE * 2);
    f16* kb   = (f16*)alloc((size_t)MTOT * EE * 2);
    f16* vb   = (f16*)alloc((size_t)MTOT * EE * 2);
    f16* ob   = (f16*)alloc((size_t)MTOT * EE * 2);
    size_t vT_bytes = (size_t)BB * HH * DHEAD * TPAD * 2;
    f16* vT   = (f16*)alloc(vT_bytes);
    f16* y1   = (f16*)alloc((size_t)MTOT * EE * XX * 2);
    f16* Wqt  = (f16*)alloc((size_t)LL * EE * EE * 2);
    f16* Wkt  = (f16*)alloc((size_t)LL * EE * EE * 2);
    f16* Wvt  = (f16*)alloc((size_t)LL * EE * EE * 2);
    f16* Wot  = (f16*)alloc((size_t)LL * EE * EE * 2);
    f16* W1t  = (f16*)alloc((size_t)LL * EE * EE * XX * 2);
    f16* W2t  = (f16*)alloc((size_t)LL * EE * EE * XX * 2);

    // weight f32 -> f16 transposed copies
    const int nEE = EE * EE, nEX = EE * EE * XX;
    for (int i = 0; i < LL; ++i) {
        wt_k<<<(nEE + 255) / 256, 256, 0, stream>>>(Wq + (size_t)i * nEE, Wqt + (size_t)i * nEE, EE, EE);
        wt_k<<<(nEE + 255) / 256, 256, 0, stream>>>(Wk + (size_t)i * nEE, Wkt + (size_t)i * nEE, EE, EE);
        wt_k<<<(nEE + 255) / 256, 256, 0, stream>>>(Wv + (size_t)i * nEE, Wvt + (size_t)i * nEE, EE, EE);
        wt_k<<<(nEE + 255) / 256, 256, 0, stream>>>(Wo + (size_t)i * nEE, Wot + (size_t)i * nEE, EE, EE);
        wt_k<<<(nEX + 255) / 256, 256, 0, stream>>>(W1 + (size_t)i * nEX, W1t + (size_t)i * nEX, EE, EE * XX);
        wt_k<<<(nEX + 255) / 256, 256, 0, stream>>>(W2 + (size_t)i * nEX, W2t + (size_t)i * nEX, EE * XX, EE);
    }
    (void)hipMemsetAsync(vT, 0, vT_bytes, stream);

    embed_k<<<BB * TT, EE, 0, stream>>>(x, W_emb, b_emb, g_emb, be_emb, clsT, h);

    dim3 g256((MTOT + 63) / 64, EE / 64);
    dim3 g1024((MTOT + 63) / 64, (EE * XX) / 64);
    for (int i = 0; i < LL; ++i) {
        ln_k<<<MTOT, EE, 0, stream>>>(h, pos, ln1_g + (size_t)i * EE, ln1_b + (size_t)i * EE, y);
        gemm_wmma<<<g256, 256, 0, stream>>>(y, Wqt + (size_t)i * nEE, bq + (size_t)i * EE,
                                            qb, nullptr, nullptr, MTOT, EE, EE, 0);
        gemm_wmma<<<g256, 256, 0, stream>>>(y, Wkt + (size_t)i * nEE, bk + (size_t)i * EE,
                                            kb, nullptr, nullptr, MTOT, EE, EE, 0);
        gemm_wmma<<<g256, 256, 0, stream>>>(y, Wvt + (size_t)i * nEE, bv_p + (size_t)i * EE,
                                            vb, nullptr, vT, MTOT, EE, EE, 3);
        attn_wmma<<<BB * HH * QT, 32, 0, stream>>>(qb, kb, vT, ob);
        gemm_wmma<<<g256, 256, 0, stream>>>(ob, Wot + (size_t)i * nEE, bo + (size_t)i * EE,
                                            h, h, nullptr, MTOT, EE, EE, 1);
        ln_k<<<MTOT, EE, 0, stream>>>(h, nullptr, ln2_g + (size_t)i * EE, ln2_b + (size_t)i * EE, y);
        gemm_wmma<<<g1024, 256, 0, stream>>>(y, W1t + (size_t)i * nEX, b1 + (size_t)i * EE * XX,
                                             y1, nullptr, nullptr, MTOT, EE * XX, EE, 2);
        gemm_wmma<<<g256, 256, 0, stream>>>(y1, W2t + (size_t)i * nEX, b2 + (size_t)i * EE,
                                            h, h, nullptr, MTOT, EE, EE * XX, 1);
    }
    cls_k<<<BB, EE, 0, stream>>>(h, Wc1, bc1, lnc_g, lnc_b, Wc2, bc2, (float*)d_out);
}
